// S4_67834713473247
// MI455X (gfx1250) — matrix operand order repaired and checked by reading, not compiled, not run on hardware
//
#include <hip/hip_runtime.h>

// ---------------------------------------------------------------------------
// S4 forward on MI455X (gfx1250):
//   y = causal_conv(u, K) + D*u  ==  Y = U @ ToeplitzLower(K) + D*U
// Implicit-Toeplitz bf16 WMMA GEMM (M=8192, N=K=4096), K built on device.
// A tiles double-buffered into LDS via the Tensor Data Mover (TENSORcnt),
// falling back to GLOBAL_LOAD_ASYNC_TO_LDS (ASYNCcnt), then plain loads.
// K held reversed in LDS so B-fragment gathers are contiguous ascending and
// causality comes from the zero-padded tail.
// ---------------------------------------------------------------------------

typedef __attribute__((ext_vector_type(16))) __bf16 v16bf;
typedef __attribute__((ext_vector_type(8)))  float  v8f;

#define N_STATE 64
#define SEQ_L   4096
#define NSEQ    (16 * 512)     // 8192 sequences
#define TILE_M  128
#define TILE_N  128
#define TILE_K  32
#define A_STRIDE 36            // fp32 elems per LDS A row (pad: conflict-free)
#define KREV_LEN (SEQ_L + TILE_N)   // 4224: zero tail encodes causal mask

typedef __attribute__((address_space(3))) void lvoid3;

#if __has_builtin(__builtin_amdgcn_tensor_load_to_lds)
#define USE_TDM 1
typedef unsigned int tu4 __attribute__((ext_vector_type(4)));
typedef int          ts4 __attribute__((ext_vector_type(4)));
typedef int          ts8 __attribute__((ext_vector_type(8)));
#else
#define USE_TDM 0
#endif

#if __has_builtin(__builtin_amdgcn_global_load_async_to_lds_b128)
#define HAS_ASYNC_LDS 1
typedef int v4i_ __attribute__((vector_size(16)));
typedef __attribute__((address_space(1))) v4i_ gv4i;
typedef __attribute__((address_space(3))) v4i_ lv4i;
#else
#define HAS_ASYNC_LDS 0
#endif

__device__ __forceinline__ void wait_async0() {
#if HAS_ASYNC_LDS
#if __has_builtin(__builtin_amdgcn_s_wait_asynccnt)
  __builtin_amdgcn_s_wait_asynccnt(0);
#else
  asm volatile("s_wait_asynccnt 0x0" ::: "memory");
#endif
#endif
}

// ---------------- Kernel 1: B = Vc@Bp and the four Cauchy numerators -------
__global__ void s4_prep_vectors(const float* __restrict__ Vc_ri,
                                const float* __restrict__ Bp,
                                const float* __restrict__ Ct_ri,
                                const float* __restrict__ p_ri,
                                const float* __restrict__ q_ri,
                                float* __restrict__ vv /* 4 x 64 x 2 */) {
  const int n = threadIdx.x;            // 64 threads
  float br = 0.f, bi = 0.f;
  for (int m = 0; m < N_STATE; ++m) {
    const float vr = Vc_ri[(n * N_STATE + m) * 2 + 0];
    const float vi = Vc_ri[(n * N_STATE + m) * 2 + 1];
    const float b  = Bp[m];
    br += vr * b; bi += vi * b;
  }
  const float cr = Ct_ri[n * 2 + 0], ci = Ct_ri[n * 2 + 1];
  const float pr = p_ri[n * 2 + 0],  pi = p_ri[n * 2 + 1];
  const float qr = q_ri[n * 2 + 0],  qi = q_ri[n * 2 + 1];
  vv[(0 * 64 + n) * 2 + 0] = cr * br + ci * bi;   // conj(Ct)*B
  vv[(0 * 64 + n) * 2 + 1] = cr * bi - ci * br;
  vv[(1 * 64 + n) * 2 + 0] = cr * pr + ci * pi;   // conj(Ct)*p
  vv[(1 * 64 + n) * 2 + 1] = cr * pi - ci * pr;
  vv[(2 * 64 + n) * 2 + 0] = qr * br + qi * bi;   // conj(q)*B
  vv[(2 * 64 + n) * 2 + 1] = qr * bi - qi * br;
  vv[(3 * 64 + n) * 2 + 0] = qr * pr + qi * pi;   // conj(q)*p
  vv[(3 * 64 + n) * 2 + 1] = qr * pi - qi * pr;
}

// ---------------- Kernel 2: generating function at the L roots of unity ----
__global__ void s4_atroots(const float* __restrict__ Lambda_ri,
                           const float* __restrict__ vv,
                           const float* __restrict__ log_step,
                           float* __restrict__ at /* 4096 x 2 */) {
  const int w = blockIdx.x * blockDim.x + threadIdx.x;
  if (w >= SEQ_L) return;
  const float step = __expf(log_step[0]);
  const float th = 6.2831853071795864769f * ((float)w / (float)SEQ_L);
  float sth, cth;
  __sincosf(th, &sth, &cth);
  const float br = 1.0f + cth, bi = -sth;         // 1 + Omega
  const float ar = 1.0f - cth, ai =  sth;         // 1 - Omega
  const float ib2 = 1.0f / (br * br + bi * bi);
  const float sc  = (2.0f / step) * ib2;
  const float gr = (ar * br + ai * bi) * sc;      // g
  const float gi = (ai * br - ar * bi) * sc;
  const float c_r =  2.0f * br * ib2;             // c = 2/(1+Omega)
  const float c_i = -2.0f * bi * ib2;
  float k00r = 0, k00i = 0, k01r = 0, k01i = 0;
  float k10r = 0, k10i = 0, k11r = 0, k11i = 0;
  for (int n = 0; n < N_STATE; ++n) {
    const float dr  = gr - Lambda_ri[n * 2 + 0];
    const float di  = gi - Lambda_ri[n * 2 + 1];
    const float idn = 1.0f / (dr * dr + di * di);
    const float ir  =  dr * idn;
    const float ii  = -di * idn;
    float vr, vi;
    vr = vv[(0 * 64 + n) * 2 + 0]; vi = vv[(0 * 64 + n) * 2 + 1];
    k00r += vr * ir - vi * ii; k00i += vr * ii + vi * ir;
    vr = vv[(1 * 64 + n) * 2 + 0]; vi = vv[(1 * 64 + n) * 2 + 1];
    k01r += vr * ir - vi * ii; k01i += vr * ii + vi * ir;
    vr = vv[(2 * 64 + n) * 2 + 0]; vi = vv[(2 * 64 + n) * 2 + 1];
    k10r += vr * ir - vi * ii; k10i += vr * ii + vi * ir;
    vr = vv[(3 * 64 + n) * 2 + 0]; vi = vv[(3 * 64 + n) * 2 + 1];
    k11r += vr * ir - vi * ii; k11i += vr * ii + vi * ir;
  }
  const float dbr = 1.0f + k11r, dbi = k11i;
  const float ib  = 1.0f / (dbr * dbr + dbi * dbi);
  const float nr  = k01r * k10r - k01i * k10i;
  const float ni  = k01r * k10i + k01i * k10r;
  const float tr  = (nr * dbr + ni * dbi) * ib;
  const float ti  = (ni * dbr - nr * dbi) * ib;
  const float kr  = k00r - tr, ki = k00i - ti;
  at[w * 2 + 0] = c_r * kr - c_i * ki;
  at[w * 2 + 1] = c_r * ki + c_i * kr;
}

// ---------------- Kernel 3: K[t] = Re(ifft(at_roots)) ----------------------
__global__ void s4_idft(const float* __restrict__ at, float* __restrict__ Kf) {
  const int t = blockIdx.x * blockDim.x + threadIdx.x;
  if (t >= SEQ_L) return;
  float acc = 0.f;
  for (int w = 0; w < SEQ_L; ++w) {
    const int r = (w * t) & (SEQ_L - 1);
    const float ph = (6.2831853071795864769f / (float)SEQ_L) * (float)r;
    float s, c;
    __sincosf(ph, &s, &c);
    acc += at[w * 2 + 0] * c - at[w * 2 + 1] * s;
  }
  Kf[t] = acc * (1.0f / (float)SEQ_L);
}

// ---------------- Kernel 4: implicit-Toeplitz WMMA GEMM --------------------
// Grid (SEQ_L/128, NSEQ/128), 256 threads = 8 waves as 2(M) x 4(N).
// Each wave: 64x32 output = 4(mi) x 2(ni) WMMA f32_16x16x32_bf16 accums.
__global__ __launch_bounds__(256)
void s4_conv_wmma(const float* __restrict__ u, const float* __restrict__ Kf,
                  const float* __restrict__ D, float* __restrict__ y) {
  __shared__ __bf16 Krev[KREV_LEN];                 // reversed K + zero tail
  __shared__ float  AuF[2][TILE_M][A_STRIDE];       // fp32 A tiles (2 bufs)

  const int tid   = threadIdx.x;
  const int lane  = tid & 31;
  const int wave  = tid >> 5;                       // 0..7
  const int wm    = wave >> 2;                      // 0..1 -> M offset *64
  const int wn    = wave & 3;                       // 0..3 -> N offset *32
  const int lrow  = lane & 15;
  const int lhalf = lane >> 4;

  const int t0 = blockIdx.x * TILE_N;               // time tile
  const int s0 = blockIdx.y * TILE_M;               // sequence tile

  // Reversed bf16 K into LDS; tail zeros realize the causal mask.
  for (int i = tid; i < SEQ_L; i += 256)
    Krev[i] = (__bf16)Kf[SEQ_L - 1 - i];
  for (int i = SEQ_L + tid; i < KREV_LEN; i += 256)
    Krev[i] = (__bf16)0.0f;

  // ---- A-tile staging: 128 rows x 32 fp32 (row stride padded to 36) ------
  auto stage_issue = [&](int kb, int buf) {
    const int tau0 = kb * TILE_K;
#if USE_TDM
    // One TDM descriptor per k-block, issued by wave 0 only (TENSORcnt).
    if (wave == 0) {
      const unsigned long long ga =
          (unsigned long long)(u + (size_t)s0 * SEQ_L + tau0);
      const unsigned lds = (unsigned)(unsigned long)(lvoid3*)&AuF[buf][0][0];
      // D# group0 (ISA 8.3): count=1 | lds_addr | global_addr | type=2
      tu4 g0;
      g0.x = 1u;                                    // count=1, user mode
      g0.y = lds;                                   // lds_addr (bytes)
      g0.z = (unsigned)ga;                          // global_addr[31:0]
      g0.w = (unsigned)((ga >> 32) & 0x1FFFFFFull) | (2u << 30); // +type=2
      // D# group1 (ISA 8.4): data_size=4B, pad every 32 DW by 4 DW,
      // tensor_dim0=4096, tensor_dim1=8192, tile=32x128, dim0_stride=4096
      ts8 g1;
      g1.s0 = (int)((2u << 16)        // data_size: 2 -> 4 bytes
                  | (1u << 20)        // pad_enable
                  | (4u << 22)        // pad_interval: 2^4*8B = 128B rows
                  | (3u << 25));      // pad_amount: 4 DWORDs (16B)
      g1.s1 = (int)((SEQ_L & 0xFFFF) << 16);        // tensor_dim0 lo16
      g1.s2 = (int)(((SEQ_L >> 16) & 0xFFFF)        // tensor_dim0 hi16
                  | ((NSEQ & 0xFFFF) << 16));       // tensor_dim1 lo16
      g1.s3 = (int)(((NSEQ >> 16) & 0xFFFF)         // tensor_dim1 hi16
                  | (TILE_K << 16));                // tile_dim0 = 32
      g1.s4 = TILE_M;                               // tile_dim1=128, dim2=0
      g1.s5 = SEQ_L;                                // dim0_stride lo32
      g1.s6 = 0;                                    // stride hi / dim1_stride
      g1.s7 = 0;
      const ts4 z4 = {0, 0, 0, 0};
#if __clang_major__ >= 23
      const ts8 z8 = {0, 0, 0, 0, 0, 0, 0, 0};
      __builtin_amdgcn_tensor_load_to_lds(g0, g1, z4, z4, z8, 0);
#else
      __builtin_amdgcn_tensor_load_to_lds(g0, g1, z4, z4, 0);
#endif
    }
#elif HAS_ASYNC_LDS
#pragma unroll
    for (int i = 0; i < 4; ++i) {
      const int chunk = tid + i * 256;
      const int row = chunk >> 3;                   // 0..127
      const int q   = chunk & 7;                    // 0..7 (float4 col)
      const float* src = u + (size_t)(s0 + row) * SEQ_L + tau0 + q * 4;
      float* dst = &AuF[buf][row][q * 4];
      __builtin_amdgcn_global_load_async_to_lds_b128(
          (gv4i*)src, (lv4i*)dst, 0, 0);
    }
#else
#pragma unroll
    for (int i = 0; i < 4; ++i) {
      const int chunk = tid + i * 256;
      const int row = chunk >> 3;
      const int q   = chunk & 7;
      const float* src = u + (size_t)(s0 + row) * SEQ_L + tau0 + q * 4;
      *(float4*)&AuF[buf][row][q * 4] = *(const float4*)src;
      __builtin_prefetch(src + TILE_K, 0, 1);       // global_prefetch_b8
    }
#endif
  };

  auto stage_wait = [&]() {
#if USE_TDM
    if (wave == 0) __builtin_amdgcn_s_wait_tensorcnt(0);
#elif HAS_ASYNC_LDS
    wait_async0();
#endif
  };

  const v8f vzero = {0.f, 0.f, 0.f, 0.f, 0.f, 0.f, 0.f, 0.f};
  v8f acc[4][2];
#pragma unroll
  for (int mi = 0; mi < 4; ++mi)
#pragma unroll
    for (int ni = 0; ni < 2; ++ni) acc[mi][ni] = vzero;

  // Triangular k-loop: tau0 <= t0 + TILE_N - 1.
  const int nkb = t0 / TILE_K + TILE_N / TILE_K;
  stage_issue(0, 0);

  for (int kb = 0; kb < nkb; ++kb) {
    const int tau0 = kb * TILE_K;
    const int buf  = kb & 1;

    stage_wait();
    __syncthreads();                                // tile visible to all waves
    if (kb + 1 < nkb) stage_issue(kb + 1, buf ^ 1);

    // A fragments: ISA 16-bit A 16x32 layout; per lane two contiguous
    // 8-float runs -> ds_load_b128 pairs, then cvt fp32->bf16.
    v16bf afrag[4];
#pragma unroll
    for (int mi = 0; mi < 4; ++mi) {
      const int m = wm * 64 + mi * 16 + lrow;
      const float4* rp = (const float4*)&AuF[buf][m][0];
      const float4 r0 = rp[lhalf * 2 + 0];          // k = lhalf*8 + 0..3
      const float4 r1 = rp[lhalf * 2 + 1];          // k = lhalf*8 + 4..7
      const float4 r2 = rp[lhalf * 2 + 4];          // k = 16+lhalf*8 + 0..3
      const float4 r3 = rp[lhalf * 2 + 5];          // k = 16+lhalf*8 + 4..7
      v16bf a;
      a[0]  = (__bf16)r0.x; a[1]  = (__bf16)r0.y;
      a[2]  = (__bf16)r0.z; a[3]  = (__bf16)r0.w;
      a[4]  = (__bf16)r1.x; a[5]  = (__bf16)r1.y;
      a[6]  = (__bf16)r1.z; a[7]  = (__bf16)r1.w;
      a[8]  = (__bf16)r2.x; a[9]  = (__bf16)r2.y;
      a[10] = (__bf16)r2.z; a[11] = (__bf16)r2.w;
      a[12] = (__bf16)r3.x; a[13] = (__bf16)r3.y;
      a[14] = (__bf16)r3.z; a[15] = (__bf16)r3.w;
      afrag[mi] = a;
    }

    // B fragments: B[k][n] = K[(t0+n)-(tau0+k)] = Krev[4095-tg+tau0+k],
    // ascending contiguous; zero tail handles d<0 (causality).
    v16bf bfrag[2];
#pragma unroll
    for (int ni = 0; ni < 2; ++ni) {
      const int tg = t0 + wn * 32 + ni * 16 + lrow;
      const int jbase = (SEQ_L - 1) - tg + tau0 + lhalf * 8;
      v16bf b;
#pragma unroll
      for (int e = 0; e < 16; ++e) b[e] = Krev[jbase + e + (e >> 3) * 8];
      bfrag[ni] = b;
    }

#pragma unroll
    for (int mi = 0; mi < 4; ++mi)
#pragma unroll
      for (int ni = 0; ni < 2; ++ni)
        acc[mi][ni] = __builtin_amdgcn_wmma_f32_16x16x32_bf16(
            false, afrag[mi], false, bfrag[ni], (short)0, acc[mi][ni],
            false, false);

    __syncthreads();                                // done reading buf
  }

  // Epilogue: y = acc + D*u.  C layout: VGPR r -> M = r + lhalf*8, N = lrow.
  const float d0 = D[0];
#pragma unroll
  for (int mi = 0; mi < 4; ++mi) {
#pragma unroll
    for (int ni = 0; ni < 2; ++ni) {
      const int tg = t0 + wn * 32 + ni * 16 + lrow;
#pragma unroll
      for (int r = 0; r < 8; ++r) {
        const int m = r + (lhalf << 3);
        const size_t idx =
            (size_t)(s0 + wm * 64 + mi * 16 + m) * SEQ_L + tg;
        y[idx] = acc[mi][ni][r] + d0 * u[idx];
      }
    }
  }
}

// ---------------------------------------------------------------------------
extern "C" void kernel_launch(void* const* d_in, const int* in_sizes, int n_in,
                              void* d_out, int out_size, void* d_ws,
                              size_t ws_size, hipStream_t stream) {
  const float* u        = (const float*)d_in[0];
  const float* Lambda   = (const float*)d_in[1];
  const float* p_ri     = (const float*)d_in[2];
  const float* q_ri     = (const float*)d_in[3];
  const float* Vc_ri    = (const float*)d_in[4];
  const float* Ct_ri    = (const float*)d_in[5];
  const float* Bp       = (const float*)d_in[6];
  const float* D        = (const float*)d_in[7];
  const float* log_step = (const float*)d_in[8];
  float* y = (float*)d_out;

  char*  ws = (char*)d_ws;
  float* vv = (float*)(ws);            //  2 KB: 4 x 64 complex numerators
  float* at = (float*)(ws + 4096);     // 32 KB: at_roots (4096 complex)
  float* Kf = (float*)(ws + 65536);    // 16 KB: conv kernel K (fp32)

  s4_prep_vectors<<<1, 64, 0, stream>>>(Vc_ri, Bp, Ct_ri, p_ri, q_ri, vv);
  s4_atroots<<<SEQ_L / 256, 256, 0, stream>>>(Lambda, vv, log_step, at);
  s4_idft<<<SEQ_L / 256, 256, 0, stream>>>(at, Kf);

  dim3 grid(SEQ_L / TILE_N, NSEQ / TILE_M);   // (32, 64)
  s4_conv_wmma<<<grid, 256, 0, stream>>>(u, Kf, D, y);
}
